// CrossModalTokenMerger_13932873908353
// MI455X (gfx1250) — matrix operand order repaired and checked by reading, not compiled, not run on hardware
//
#include <hip/hip_runtime.h>
#include <math.h>

// ---------------------------------------------------------------------------
// CrossModalTokenMerger for MI455X (gfx1250, wave32).
//
// Roofline: ~280MB HBM traffic (~12us @ 23.3TB/s) vs 103 GFLOP of GEMM.
// fp32 WMMA (16x16x4) would be compute-bound (~170us+), so all three GEMMs
// run on V_WMMA_F32_16X16X32_F16 with split-f16 error correction:
//   x = hi + lo (hi=f16(x), lo=f16(x-hi));  D += ah*bh + ah*bl + al*bh
// => ~22 mantissa bits (~1e-7 rel, fp32-equivalent for the sim>0.7 test) at
// 3 WMMAs / 32-k chunk. The hi/lo split is done ONCE per element in
// elementwise pre-passes; GEMM inner loops are pure b128 f16 loads + WMMA.
// Waves compute 32x64 output tiles (2 A fragments x 4 B tiles, 8 f32
// accumulators) to halve B-operand L2 traffic: 24 b128 loads per 24 WMMAs.
// ---------------------------------------------------------------------------

typedef float    v8f  __attribute__((ext_vector_type(8)));
typedef _Float16 v16h __attribute__((ext_vector_type(16)));
typedef _Float16 v8h  __attribute__((ext_vector_type(8)));
typedef _Float16 v4h  __attribute__((ext_vector_type(4)));

#define B_     8
#define L_     1024
#define V_     1024
#define D_     2048
#define DV_    1024
#define K_     512          // max merged pairs per batch
#define H_     1024         // MLP hidden (D/2)
#define CAND_  2048         // candidate cap (expected ~770 over threshold)
#define ML_    1536         // padded output length (L + K)
#define TEMP_   0.1f
#define THRESH_ 0.7f

static __device__ __forceinline__ v8f wmma32(v16h a, v16h b, v8f c) {
  // D(16x16,f32) = A(16x32,f16) * B(32x16,f16) + C
  return __builtin_amdgcn_wmma_f32_16x16x32_f16(false, a, false, b, (short)0, c,
                                                false, false);
}

static __device__ __forceinline__ v8f mma3(v16h ah, v16h al, v16h bh, v16h bl,
                                           v8f c) {
  c = wmma32(ah, bh, c);
  c = wmma32(ah, bl, c);
  c = wmma32(al, bh, c);
  return c;
}

// A fragment: elems 0..15 from p[0..15]
static __device__ __forceinline__ v16h loadA(const _Float16* __restrict__ p) {
  const v8h a = *(const v8h*)(p);
  const v8h b = *(const v8h*)(p + 8);
  return __builtin_shufflevector(a, b, 0, 1, 2, 3, 4, 5, 6, 7, 8, 9, 10, 11,
                                 12, 13, 14, 15);
}
// B fragment: 16 contiguous K values of the B^T row
static __device__ __forceinline__ v16h loadB16(const _Float16* __restrict__ p) {
  const v8h a = *(const v8h*)(p);
  const v8h b = *(const v8h*)(p + 8);
  return __builtin_shufflevector(a, b, 0, 1, 2, 3, 4, 5, 6, 7, 8, 9, 10, 11,
                                 12, 13, 14, 15);
}

// ---------------- elementwise fp32 -> (hi,lo) f16 split ----------------------
__global__ void k_split(const float* __restrict__ src, _Float16* __restrict__ hi,
                        _Float16* __restrict__ lo, int n4) {
  const int i = blockIdx.x * blockDim.x + threadIdx.x;
  if (i >= n4) return;
  const float4 x = ((const float4*)src)[i];
  v4h h, l;
  h[0] = (_Float16)x.x; l[0] = (_Float16)(x.x - (float)h[0]);
  h[1] = (_Float16)x.y; l[1] = (_Float16)(x.y - (float)h[1]);
  h[2] = (_Float16)x.z; l[2] = (_Float16)(x.z - (float)h[2]);
  h[3] = (_Float16)x.w; l[3] = (_Float16)(x.w - (float)h[3]);
  ((v4h*)hi)[i] = h;
  ((v4h*)lo)[i] = l;
}

// ---------------- GEMM 1: vis_proj = vision @ W_v2l^T + bias (split out) ----
// grid (8, 32, 8) block (32,4); each wave -> 32x64 tile of [V, D].
__global__ void k_visproj(const _Float16* __restrict__ vish,
                          const _Float16* __restrict__ visl,
                          const _Float16* __restrict__ Wh,
                          const _Float16* __restrict__ Wl,
                          const float* __restrict__ bias,
                          _Float16* __restrict__ vph,
                          _Float16* __restrict__ vpl) {
  const int lane = threadIdx.x;
  const int b    = blockIdx.z;
  const int mt   = blockIdx.y;                                // 32-row tile
  const int ng   = blockIdx.x * blockDim.y + threadIdx.y;     // 0..31, 64 cols
  const int m0   = (mt << 5) + (lane & 15);
  const int n0   = (ng << 6) + (lane & 15);
  const int aoff = (lane >> 4) << 4;                          // 0 or 16
  const int boff = (lane >> 4) << 4;                          // 0 or 16
  const _Float16* ah0p = vish + ((size_t)b * V_ + m0) * DV_;
  const _Float16* al0p = visl + ((size_t)b * V_ + m0) * DV_;
  const _Float16* ah1p = ah0p + (size_t)16 * DV_;
  const _Float16* al1p = al0p + (size_t)16 * DV_;
  const _Float16* bh0 = Wh + (size_t)(n0 +  0) * DV_;
  const _Float16* bh1 = Wh + (size_t)(n0 + 16) * DV_;
  const _Float16* bh2 = Wh + (size_t)(n0 + 32) * DV_;
  const _Float16* bh3 = Wh + (size_t)(n0 + 48) * DV_;
  const _Float16* bl0 = Wl + (size_t)(n0 +  0) * DV_;
  const _Float16* bl1 = Wl + (size_t)(n0 + 16) * DV_;
  const _Float16* bl2 = Wl + (size_t)(n0 + 32) * DV_;
  const _Float16* bl3 = Wl + (size_t)(n0 + 48) * DV_;
  v8f c0 = {}, c1 = {}, c2 = {}, c3 = {};
  v8f c4 = {}, c5 = {}, c6 = {}, c7 = {};
  for (int kc = 0; kc < DV_; kc += 32) {
    if ((kc & 255) == 0) {
      __builtin_prefetch(ah0p + kc + 256, 0, 0);
      __builtin_prefetch(ah1p + kc + 256, 0, 0);
    }
    const v16h ah0 = loadA(ah0p + kc + aoff);
    const v16h al0 = loadA(al0p + kc + aoff);
    const v16h ah1 = loadA(ah1p + kc + aoff);
    const v16h al1 = loadA(al1p + kc + aoff);
    v16h bh, bl;
    bh = loadB16(bh0 + kc + boff); bl = loadB16(bl0 + kc + boff);
    c0 = mma3(ah0, al0, bh, bl, c0); c4 = mma3(ah1, al1, bh, bl, c4);
    bh = loadB16(bh1 + kc + boff); bl = loadB16(bl1 + kc + boff);
    c1 = mma3(ah0, al0, bh, bl, c1); c5 = mma3(ah1, al1, bh, bl, c5);
    bh = loadB16(bh2 + kc + boff); bl = loadB16(bl2 + kc + boff);
    c2 = mma3(ah0, al0, bh, bl, c2); c6 = mma3(ah1, al1, bh, bl, c6);
    bh = loadB16(bh3 + kc + boff); bl = loadB16(bl3 + kc + boff);
    c3 = mma3(ah0, al0, bh, bl, c3); c7 = mma3(ah1, al1, bh, bl, c7);
  }
  const int rowbase = (mt << 5) + ((lane >> 4) << 3);
  const int colbase = (ng << 6) + (lane & 15);
  _Float16* oh = vph + (size_t)b * V_ * D_;
  _Float16* ol = vpl + (size_t)b * V_ * D_;
  v8f accs[8] = {c0, c1, c2, c3, c4, c5, c6, c7};
#pragma unroll
  for (int t = 0; t < 8; ++t)
#pragma unroll
    for (int r = 0; r < 8; ++r) {
      const int row = rowbase + ((t >> 2) << 4) + r;
      const int col = colbase + ((t & 3) << 4);
      const float val = accs[t][r] + bias[col];
      const _Float16 h = (_Float16)val;
      oh[(size_t)row * D_ + col] = h;
      ol[(size_t)row * D_ + col] = (_Float16)(val - (float)h);
    }
}

// ---------------- row L2 norms (clamped), one wave per row -------------------
__global__ void k_rownorm(const float* __restrict__ src, float* __restrict__ dst,
                          int nrows) {
  const int lane = threadIdx.x;
  const int row  = blockIdx.x * blockDim.y + threadIdx.y;
  if (row >= nrows) return;
  const float* p = src + (size_t)row * D_;
  float s = 0.f;
  for (int i = lane; i < D_; i += 32) { float x = p[i]; s += x * x; }
#pragma unroll
  for (int off = 16; off; off >>= 1) s += __shfl_xor(s, off, 32);
  if (lane == 0) dst[row] = fmaxf(sqrtf(s), 1e-12f);
}

// norm of a split (hi+lo) matrix row
__global__ void k_rownorm_hl(const _Float16* __restrict__ hi,
                             const _Float16* __restrict__ lo,
                             float* __restrict__ dst, int nrows) {
  const int lane = threadIdx.x;
  const int row  = blockIdx.x * blockDim.y + threadIdx.y;
  if (row >= nrows) return;
  const _Float16* ph = hi + (size_t)row * D_;
  const _Float16* pl = lo + (size_t)row * D_;
  float s = 0.f;
  for (int i = lane; i < D_; i += 32) {
    const float x = (float)ph[i] + (float)pl[i];
    s += x * x;
  }
#pragma unroll
  for (int off = 16; off; off >>= 1) s += __shfl_xor(s, off, 32);
  if (lane == 0) dst[row] = fmaxf(sqrtf(s), 1e-12f);
}

// ---------------- GEMM 2: sim = (lang @ vis_proj^T)/(nl*nv*TEMP) -------------
// grid (4, 32, 8) block (32,4); wave -> 32x64 tile. Epilogue scatters
// >THRESH candidates.
__global__ void k_sim(const _Float16* __restrict__ langh,
                      const _Float16* __restrict__ langl,
                      const _Float16* __restrict__ vph,
                      const _Float16* __restrict__ vpl,
                      const float* __restrict__ nl, const float* __restrict__ nv,
                      float* __restrict__ sim, float* __restrict__ cval,
                      int* __restrict__ cli, int* __restrict__ cvi,
                      int* __restrict__ cnt) {
  const int lane = threadIdx.x;
  const int b    = blockIdx.z;
  const int mt   = blockIdx.y;                                // 32-row tile
  const int ng   = blockIdx.x * blockDim.y + threadIdx.y;     // 0..15
  const int m0   = (mt << 5) + (lane & 15);
  const int n0   = (ng << 6) + (lane & 15);
  const int aoff = (lane >> 4) << 4;
  const int boff = (lane >> 4) << 4;
  const _Float16* ah0p = langh + ((size_t)b * L_ + m0) * D_;
  const _Float16* al0p = langl + ((size_t)b * L_ + m0) * D_;
  const _Float16* ah1p = ah0p + (size_t)16 * D_;
  const _Float16* al1p = al0p + (size_t)16 * D_;
  const _Float16* bh0 = vph + ((size_t)b * V_ + n0 +  0) * D_;
  const _Float16* bh1 = vph + ((size_t)b * V_ + n0 + 16) * D_;
  const _Float16* bh2 = vph + ((size_t)b * V_ + n0 + 32) * D_;
  const _Float16* bh3 = vph + ((size_t)b * V_ + n0 + 48) * D_;
  const _Float16* bl0 = vpl + ((size_t)b * V_ + n0 +  0) * D_;
  const _Float16* bl1 = vpl + ((size_t)b * V_ + n0 + 16) * D_;
  const _Float16* bl2 = vpl + ((size_t)b * V_ + n0 + 32) * D_;
  const _Float16* bl3 = vpl + ((size_t)b * V_ + n0 + 48) * D_;
  v8f c0 = {}, c1 = {}, c2 = {}, c3 = {};
  v8f c4 = {}, c5 = {}, c6 = {}, c7 = {};
  for (int kc = 0; kc < D_; kc += 32) {
    if ((kc & 255) == 0) {
      __builtin_prefetch(ah0p + kc + 256, 0, 0);
      __builtin_prefetch(ah1p + kc + 256, 0, 0);
    }
    const v16h ah0 = loadA(ah0p + kc + aoff);
    const v16h al0 = loadA(al0p + kc + aoff);
    const v16h ah1 = loadA(ah1p + kc + aoff);
    const v16h al1 = loadA(al1p + kc + aoff);
    v16h bh, bl;
    bh = loadB16(bh0 + kc + boff); bl = loadB16(bl0 + kc + boff);
    c0 = mma3(ah0, al0, bh, bl, c0); c4 = mma3(ah1, al1, bh, bl, c4);
    bh = loadB16(bh1 + kc + boff); bl = loadB16(bl1 + kc + boff);
    c1 = mma3(ah0, al0, bh, bl, c1); c5 = mma3(ah1, al1, bh, bl, c5);
    bh = loadB16(bh2 + kc + boff); bl = loadB16(bl2 + kc + boff);
    c2 = mma3(ah0, al0, bh, bl, c2); c6 = mma3(ah1, al1, bh, bl, c6);
    bh = loadB16(bh3 + kc + boff); bl = loadB16(bl3 + kc + boff);
    c3 = mma3(ah0, al0, bh, bl, c3); c7 = mma3(ah1, al1, bh, bl, c7);
  }
  const int rowbase = (mt << 5) + ((lane >> 4) << 3);
  const int colbase = (ng << 6) + (lane & 15);
  v8f accs[8] = {c0, c1, c2, c3, c4, c5, c6, c7};
#pragma unroll
  for (int t = 0; t < 8; ++t)
#pragma unroll
    for (int r = 0; r < 8; ++r) {
      const int row = rowbase + ((t >> 2) << 4) + r;
      const int col = colbase + ((t & 3) << 4);
      const float val =
          accs[t][r] / (nl[b * L_ + row] * nv[b * V_ + col] * TEMP_);
      sim[((size_t)b * L_ + row) * V_ + col] = val;
      if (val > THRESH_) {
        const int idx = atomicAdd(cnt + b, 1);
        if (idx < CAND_) {
          cval[b * CAND_ + idx] = val;
          cli[b * CAND_ + idx]  = row;
          cvi[b * CAND_ + idx]  = col;
        }
      }
    }
}

// ---------------- per-batch: sort candidates desc, build plans ---------------
__global__ __launch_bounds__(1024) void k_plan(
    const float* __restrict__ cval, const int* __restrict__ cli,
    const int* __restrict__ cvi, const int* __restrict__ cnt,
    int* __restrict__ pl, int* __restrict__ pv, int* __restrict__ nsel,
    int* __restrict__ lkeep, int* __restrict__ lsrc, int* __restrict__ vsrc) {
  __shared__ float sv[CAND_];
  __shared__ int   sl[CAND_], ss[CAND_];
  __shared__ int   f[1024], g[1024];
  const int tid = threadIdx.x;
  const int b   = blockIdx.x;
  int c = cnt[b]; if (c > CAND_) c = CAND_;
  for (int i = tid; i < CAND_; i += 1024) {
    if (i < c) { sv[i] = cval[b * CAND_ + i]; sl[i] = cli[b * CAND_ + i]; ss[i] = cvi[b * CAND_ + i]; }
    else       { sv[i] = -1e30f; sl[i] = 0; ss[i] = 0; }
  }
  __syncthreads();
  // bitonic sort, descending by value
  for (int ks = 2; ks <= CAND_; ks <<= 1) {
    for (int j = ks >> 1; j > 0; j >>= 1) {
      for (int i = tid; i < CAND_; i += 1024) {
        const int ixj = i ^ j;
        if (ixj > i) {
          const bool desc = ((i & ks) == 0);
          const float a = sv[i], bb = sv[ixj];
          if (desc ? (a < bb) : (a > bb)) {
            sv[i] = bb; sv[ixj] = a;
            int t1 = sl[i]; sl[i] = sl[ixj]; sl[ixj] = t1;
            int t2 = ss[i]; ss[i] = ss[ixj]; ss[ixj] = t2;
          }
        }
      }
      __syncthreads();
    }
  }
  const int n = (c < K_) ? c : K_;   // topk-then-threshold == first min(c,K)
  for (int p = tid; p < K_; p += 1024) {
    pl[b * K_ + p] = (p < n) ? sl[p] : 0;
    pv[b * K_ + p] = (p < n) ? ss[p] : 0;
  }
  if (tid == 0) nsel[b] = n;
  // ---- language keep/compact plan
  f[tid] = 1; __syncthreads();
  for (int p = tid; p < n; p += 1024) f[sl[p]] = 0;
  __syncthreads();
  g[tid] = f[tid]; __syncthreads();
  for (int off = 1; off < 1024; off <<= 1) {
    const int v = (tid >= off) ? g[tid - off] : 0;
    __syncthreads();
    g[tid] += v;
    __syncthreads();
  }
  const int Lk = g[1023];
  if (f[tid]) lsrc[b * ML_ + (g[tid] - 1)] = tid;       // kept lang row -> dest
  if (tid == 0) lkeep[b] = Lk;
  for (int r = Lk + tid; r < ML_; r += 1024)
    lsrc[b * ML_ + r] = (r < Lk + n) ? -2 : -1;          // -2 merged, -1 pad
  __syncthreads();
  // ---- vision keep/compact plan
  f[tid] = 1; __syncthreads();
  for (int p = tid; p < n; p += 1024) f[ss[p]] = 0;
  __syncthreads();
  g[tid] = f[tid]; __syncthreads();
  for (int off = 1; off < 1024; off <<= 1) {
    const int v = (tid >= off) ? g[tid - off] : 0;
    __syncthreads();
    g[tid] += v;
    __syncthreads();
  }
  const int Vk = g[1023];
  if (f[tid]) vsrc[b * ML_ + (g[tid] - 1)] = tid;        // kept vision row
  for (int r = tid; r < n; r += 1024) vsrc[b * ML_ + Vk + r] = ss[r];
  for (int r = Vk + n + tid; r < ML_; r += 1024) vsrc[b * ML_ + r] = -1;
}

// ---------------- GEMM 3: H = relu([lt|vt] @ W_m1^T + b_m1), M = 512 pairs ---
// grid (4, 16, 8) block (32,4); wave -> 32x64 tile.
__global__ void k_mlp1(const _Float16* __restrict__ langh,
                       const _Float16* __restrict__ langl,
                       const _Float16* __restrict__ vph,
                       const _Float16* __restrict__ vpl,
                       const int* __restrict__ pl, const int* __restrict__ pv,
                       const _Float16* __restrict__ W1h,
                       const _Float16* __restrict__ W1l,
                       const float* __restrict__ b1,
                       float* __restrict__ Hbuf) {
  const int lane = threadIdx.x;
  const int b    = blockIdx.z;
  const int mt   = blockIdx.y;                                // 32-pair tile
  const int ng   = blockIdx.x * blockDim.y + threadIdx.y;     // 0..15
  const int p0   = (mt << 5) + (lane & 15);
  const int n0   = (ng << 6) + (lane & 15);
  const int aoff = (lane >> 4) << 4;
  const int boff = (lane >> 4) << 4;
  const int li0 = pl[b * K_ + p0],      vi0 = pv[b * K_ + p0];
  const int li1 = pl[b * K_ + p0 + 16], vi1 = pv[b * K_ + p0 + 16];
  const _Float16* lrh0 = langh + ((size_t)b * L_ + li0) * D_;
  const _Float16* lrl0 = langl + ((size_t)b * L_ + li0) * D_;
  const _Float16* vrh0 = vph + ((size_t)b * V_ + vi0) * D_;
  const _Float16* vrl0 = vpl + ((size_t)b * V_ + vi0) * D_;
  const _Float16* lrh1 = langh + ((size_t)b * L_ + li1) * D_;
  const _Float16* lrl1 = langl + ((size_t)b * L_ + li1) * D_;
  const _Float16* vrh1 = vph + ((size_t)b * V_ + vi1) * D_;
  const _Float16* vrl1 = vpl + ((size_t)b * V_ + vi1) * D_;
  const _Float16* wh0 = W1h + (size_t)(n0 +  0) * (2 * D_);
  const _Float16* wh1 = W1h + (size_t)(n0 + 16) * (2 * D_);
  const _Float16* wh2 = W1h + (size_t)(n0 + 32) * (2 * D_);
  const _Float16* wh3 = W1h + (size_t)(n0 + 48) * (2 * D_);
  const _Float16* wl0 = W1l + (size_t)(n0 +  0) * (2 * D_);
  const _Float16* wl1 = W1l + (size_t)(n0 + 16) * (2 * D_);
  const _Float16* wl2 = W1l + (size_t)(n0 + 32) * (2 * D_);
  const _Float16* wl3 = W1l + (size_t)(n0 + 48) * (2 * D_);
  v8f c0 = {}, c1 = {}, c2 = {}, c3 = {};
  v8f c4 = {}, c5 = {}, c6 = {}, c7 = {};
  for (int kc = 0; kc < 2 * D_; kc += 32) {
    // 32-k chunk never straddles the lang|vis boundary (2048 % 32 == 0)
    const bool inL = (kc < D_);
    const int  ko  = inL ? kc : (kc - D_);
    const v16h ah0 = loadA((inL ? lrh0 : vrh0) + ko + aoff);
    const v16h al0 = loadA((inL ? lrl0 : vrl0) + ko + aoff);
    const v16h ah1 = loadA((inL ? lrh1 : vrh1) + ko + aoff);
    const v16h al1 = loadA((inL ? lrl1 : vrl1) + ko + aoff);
    v16h bh, bl;
    bh = loadB16(wh0 + kc + boff); bl = loadB16(wl0 + kc + boff);
    c0 = mma3(ah0, al0, bh, bl, c0); c4 = mma3(ah1, al1, bh, bl, c4);
    bh = loadB16(wh1 + kc + boff); bl = loadB16(wl1 + kc + boff);
    c1 = mma3(ah0, al0, bh, bl, c1); c5 = mma3(ah1, al1, bh, bl, c5);
    bh = loadB16(wh2 + kc + boff); bl = loadB16(wl2 + kc + boff);
    c2 = mma3(ah0, al0, bh, bl, c2); c6 = mma3(ah1, al1, bh, bl, c6);
    bh = loadB16(wh3 + kc + boff); bl = loadB16(wl3 + kc + boff);
    c3 = mma3(ah0, al0, bh, bl, c3); c7 = mma3(ah1, al1, bh, bl, c7);
  }
  const int rowbase = (mt << 5) + ((lane >> 4) << 3);
  const int colbase = (ng << 6) + (lane & 15);
  v8f accs[8] = {c0, c1, c2, c3, c4, c5, c6, c7};
#pragma unroll
  for (int t = 0; t < 8; ++t)
#pragma unroll
    for (int r = 0; r < 8; ++r) {
      const int row = rowbase + ((t >> 2) << 4) + r;
      const int col = colbase + ((t & 3) << 4);
      Hbuf[((size_t)b * K_ + row) * H_ + col] =
          fmaxf(accs[t][r] + b1[col], 0.f);
    }
}

// ---------------- per-pair: softmax weights + merged row (one wave/pair) -----
__global__ void k_merge(const float* __restrict__ lang,
                        const _Float16* __restrict__ vph,
                        const _Float16* __restrict__ vpl,
                        const float* __restrict__ Hbuf,
                        const float* __restrict__ W2, const float* __restrict__ b2,
                        const int* __restrict__ pl, const int* __restrict__ pv,
                        const int* __restrict__ nsel, const int* __restrict__ lkeep,
                        float* __restrict__ outL) {
  const int lane = threadIdx.x;
  const int gid  = blockIdx.x * blockDim.y + threadIdx.y;
  const int b = gid >> 9, p = gid & (K_ - 1);
  if (p >= nsel[b]) return;
  const float* h = Hbuf + ((size_t)b * K_ + p) * H_;
  float s0 = 0.f, s1 = 0.f;
  for (int i = lane; i < H_; i += 32) {
    const float x = h[i];
    s0 += W2[i] * x;
    s1 += W2[H_ + i] * x;
  }
#pragma unroll
  for (int off = 16; off; off >>= 1) {
    s0 += __shfl_xor(s0, off, 32);
    s1 += __shfl_xor(s1, off, 32);
  }
  s0 += b2[0]; s1 += b2[1];
  const float mx = fmaxf(s0, s1);
  const float e0 = expf(s0 - mx), e1 = expf(s1 - mx);
  const float w0 = e0 / (e0 + e1), w1 = e1 / (e0 + e1);
  const int li = pl[b * K_ + p], vi2 = pv[b * K_ + p];
  const float* lr = lang + ((size_t)b * L_ + li) * D_;
  const _Float16* vh = vph + ((size_t)b * V_ + vi2) * D_;
  const _Float16* vl = vpl + ((size_t)b * V_ + vi2) * D_;
  float* orow = outL + ((size_t)b * ML_ + lkeep[b] + p) * D_;
  for (int d = lane; d < D_; d += 32) {
    const float vt = (float)vh[d] + (float)vl[d];
    orow[d] = w0 * lr[d] + w1 * vt;
  }
}

// ---------------- output writers (gather kept rows / zero-pad) ---------------
__global__ void k_write_lang(const float* __restrict__ lang,
                             const int* __restrict__ lsrc,
                             float* __restrict__ outL) {
  const int b = blockIdx.x / ML_, r = blockIdx.x % ML_;
  const int s = lsrc[b * ML_ + r];
  if (s == -2) return;  // merged row: written by k_merge
  float4* dst = (float4*)(outL + ((size_t)b * ML_ + r) * D_);
  if (s >= 0) {
    const float4* srcp = (const float4*)(lang + ((size_t)b * L_ + s) * D_);
    for (int i = threadIdx.x; i < D_ / 4; i += blockDim.x) dst[i] = srcp[i];
  } else {
    const float4 z = {0.f, 0.f, 0.f, 0.f};
    for (int i = threadIdx.x; i < D_ / 4; i += blockDim.x) dst[i] = z;
  }
}

__global__ void k_write_vis(const float* __restrict__ vis,
                            const int* __restrict__ vsrc,
                            float* __restrict__ outV) {
  const int b = blockIdx.x / ML_, r = blockIdx.x % ML_;
  const int s = vsrc[b * ML_ + r];
  float4* dst = (float4*)(outV + ((size_t)b * ML_ + r) * DV_);
  if (s >= 0) {
    const float4* srcp = (const float4*)(vis + ((size_t)b * V_ + s) * DV_);
    for (int i = threadIdx.x; i < DV_ / 4; i += blockDim.x) dst[i] = srcp[i];
  } else {
    const float4 z = {0.f, 0.f, 0.f, 0.f};
    for (int i = threadIdx.x; i < DV_ / 4; i += blockDim.x) dst[i] = z;
  }
}

__global__ void k_init(int* __restrict__ cnt) {
  if (threadIdx.x < B_) cnt[threadIdx.x] = 0;
}

// ---------------------------------------------------------------------------
extern "C" void kernel_launch(void* const* d_in, const int* in_sizes, int n_in,
                              void* d_out, int out_size, void* d_ws,
                              size_t ws_size, hipStream_t stream) {
  (void)in_sizes; (void)n_in; (void)out_size; (void)ws_size;
  const float* lang = (const float*)d_in[0];
  const float* vis  = (const float*)d_in[1];
  const float* Wv   = (const float*)d_in[2];
  const float* bv   = (const float*)d_in[3];
  const float* W1   = (const float*)d_in[4];
  const float* b1   = (const float*)d_in[5];
  const float* W2   = (const float*)d_in[6];
  const float* b2   = (const float*)d_in[7];

  float* outL = (float*)d_out;                       // [8,1536,2048]
  float* outV = outL + (size_t)B_ * ML_ * D_;        // [8,1536,1024]
  float* outS = outV + (size_t)B_ * ML_ * DV_;       // [8,1024,1024]

  const size_t nLang = (size_t)B_ * L_ * D_;   // 16.7M
  const size_t nVis  = (size_t)B_ * V_ * DV_;  //  8.4M
  const size_t nWv   = (size_t)D_ * DV_;       //  2.1M
  const size_t nW1   = (size_t)H_ * 2 * D_;    //  4.2M
  const size_t nVp   = (size_t)B_ * V_ * D_;   // 16.7M

  char* w = (char*)d_ws;
  _Float16* langh = (_Float16*)w; w += nLang * 2;
  _Float16* langl = (_Float16*)w; w += nLang * 2;
  _Float16* vish  = (_Float16*)w; w += nVis * 2;
  _Float16* visl  = (_Float16*)w; w += nVis * 2;
  _Float16* Wvh   = (_Float16*)w; w += nWv * 2;
  _Float16* Wvl   = (_Float16*)w; w += nWv * 2;
  _Float16* W1h   = (_Float16*)w; w += nW1 * 2;
  _Float16* W1l   = (_Float16*)w; w += nW1 * 2;
  _Float16* vph   = (_Float16*)w; w += nVp * 2;
  _Float16* vpl   = (_Float16*)w; w += nVp * 2;
  float* Hbuf = (float*)w; w += (size_t)B_ * K_ * H_ * 4;   // 16 MB
  float* nl   = (float*)w; w += (size_t)B_ * L_ * 4;
  float* nv   = (float*)w; w += (size_t)B_ * V_ * 4;
  float* cval = (float*)w; w += (size_t)B_ * CAND_ * 4;
  int*   cli  = (int*)w;   w += (size_t)B_ * CAND_ * 4;
  int*   cvi  = (int*)w;   w += (size_t)B_ * CAND_ * 4;
  int*   cnt  = (int*)w;   w += 64;
  int*   nsel = (int*)w;   w += 64;
  int*   lkp  = (int*)w;   w += 64;
  int*   pl   = (int*)w;   w += (size_t)B_ * K_ * 4;
  int*   pv   = (int*)w;   w += (size_t)B_ * K_ * 4;
  int*   lsrc = (int*)w;   w += (size_t)B_ * ML_ * 4;
  int*   vsrc = (int*)w;   w += (size_t)B_ * ML_ * 4;

  k_init<<<dim3(1), dim3(32), 0, stream>>>(cnt);
  // one-time hi/lo splits (bandwidth-bound, O(elements))
  k_split<<<dim3((int)(nLang / 4 / 256)), dim3(256), 0, stream>>>(lang, langh, langl, (int)(nLang / 4));
  k_split<<<dim3((int)(nVis  / 4 / 256)), dim3(256), 0, stream>>>(vis, vish, visl, (int)(nVis / 4));
  k_split<<<dim3((int)(nWv   / 4 / 256)), dim3(256), 0, stream>>>(Wv, Wvh, Wvl, (int)(nWv / 4));
  k_split<<<dim3((int)(nW1   / 4 / 256)), dim3(256), 0, stream>>>(W1, W1h, W1l, (int)(nW1 / 4));

  k_visproj<<<dim3(8, 32, 8), dim3(32, 4), 0, stream>>>(vish, visl, Wvh, Wvl,
                                                        bv, vph, vpl);
  k_rownorm<<<dim3((B_ * L_) / 8), dim3(32, 8), 0, stream>>>(lang, nl, B_ * L_);
  k_rownorm_hl<<<dim3((B_ * V_) / 8), dim3(32, 8), 0, stream>>>(vph, vpl, nv,
                                                                B_ * V_);
  k_sim<<<dim3(4, 32, 8), dim3(32, 4), 0, stream>>>(langh, langl, vph, vpl, nl,
                                                    nv, outS, cval, cli, cvi,
                                                    cnt);
  k_plan<<<dim3(B_), dim3(1024), 0, stream>>>(cval, cli, cvi, cnt, pl, pv,
                                              nsel, lkp, lsrc, vsrc);
  k_mlp1<<<dim3(4, 16, 8), dim3(32, 4), 0, stream>>>(langh, langl, vph, vpl,
                                                     pl, pv, W1h, W1l, b1,
                                                     Hbuf);
  k_merge<<<dim3(B_ * K_ / 8), dim3(32, 8), 0, stream>>>(
      lang, vph, vpl, Hbuf, W2, b2, pl, pv, nsel, lkp, outL);
  k_write_lang<<<dim3(B_ * ML_), dim3(256), 0, stream>>>(lang, lsrc, outL);
  k_write_vis<<<dim3(B_ * ML_), dim3(256), 0, stream>>>(vis, vsrc, outV);
}
